// UnsupervisedTriphard_74139725464113
// MI455X (gfx1250) — compile-verified, matrix operand order checked by reading, not run on hardware
//
#include <hip/hip_runtime.h>

// Problem constants (from reference)
#define NROWS  16384
#define DDIM   512
#define MARGIN_F 2.0f
#define EPS_F    1e-6f

#define WAVES  4            // waves per workgroup
#define TILEC  32           // columns processed per iteration (2 WMMA tiles)
#define NT     (NROWS / TILEC)

typedef __attribute__((ext_vector_type(16))) _Float16 v16h;
typedef __attribute__((ext_vector_type(8)))  float    v8f;
typedef __attribute__((ext_vector_type(4)))  unsigned int v4u;

// ---------------------------------------------------------------------------
// Kernel 1: per-row squared norm + fp32 -> (f16 hi, f16 lo) split
// ---------------------------------------------------------------------------
__global__ __launch_bounds__(128) void prep_kernel(
    const float* __restrict__ x,
    _Float16* __restrict__ hi, _Float16* __restrict__ lo,
    float* __restrict__ sq)
{
  __shared__ float red[128];
  const int row = blockIdx.x;
  const int t   = threadIdx.x;
  const float* xr = x + (size_t)row * DDIM;
  _Float16* hr = hi + (size_t)row * DDIM;
  _Float16* lr = lo + (size_t)row * DDIM;
  float acc = 0.f;
  for (int k = t; k < DDIM; k += 128) {
    float v = xr[k];
    _Float16 h = (_Float16)v;          // RTNE
    float res = v - (float)h;          // exact residual
    hr[k] = h;
    lr[k] = (_Float16)res;
    acc += v * v;
  }
  red[t] = acc;
  __syncthreads();
  for (int s = 64; s > 0; s >>= 1) {
    if (t < s) red[t] += red[t + s];
    __syncthreads();
  }
  if (t == 0) sq[row] = red[0];
}

// ---------------------------------------------------------------------------
// CDNA5 async copy: per-lane 16B global -> LDS, tracked by ASYNCcnt.
// ---------------------------------------------------------------------------
__device__ inline void async_b128(unsigned lds_off, unsigned long long gaddr) {
  asm volatile("global_load_async_to_lds_b128 %0, %1, off"
               :: "v"(lds_off), "v"(gaddr) : "memory");
}
__device__ inline void wait_async0() {
  asm volatile("s_wait_asynccnt 0x0" ::: "memory");
}
__device__ inline unsigned lds_off_of(const void* p) {
  return (unsigned)(unsigned long long)p;   // generic LDS ptr: offset in low 32b
}

// Load one 16-half WMMA fragment slice from an LDS row pointer.
// Layout per ISA (16-bit A 16x32): lane<16 -> K {0..7, 16..23},
// lane>=16 -> K {8..15, 24..31}.
__device__ inline v16h ld_frag(const _Float16* p, int sel) {
  union { v4u u[2]; v16h h; } r;
  r.u[0] = *(const v4u*)(p + sel * 8);
  r.u[1] = *(const v4u*)(p + 16 + sel * 8);
  return r.h;
}

// ---------------------------------------------------------------------------
// Kernel 2: fused split-f16 Gram GEMM + streaming per-row top-3 + loss terms.
// One WG = 4 waves = 64 rows; async double-buffered 32-column B tile;
// 2 WMMA column tiles per iteration reuse the A fragments (2 ds_loads/WMMA).
// ---------------------------------------------------------------------------
__global__ __launch_bounds__(128) void triphard_main(
    const float* __restrict__ x, const float* __restrict__ pos,
    const _Float16* __restrict__ Ahi, const _Float16* __restrict__ Alo,
    const float* __restrict__ sq, float* __restrict__ lossterm)
{
  __shared__ _Float16 sAhi[WAVES][16 * DDIM];   // 64 KB
  __shared__ _Float16 sAlo[WAVES][16 * DDIM];   // 64 KB
  __shared__ _Float16 sBhi[2][TILEC * DDIM];    // 64 KB (double buffered)
  __shared__ _Float16 sBlo[2][TILEC * DDIM];    // 64 KB (double buffered)
  __shared__ float    sS[WAVES][16][TILEC + 1]; // padded C scratch
  __shared__ int      sNeg[WAVES][16];

  const int tid  = threadIdx.x;
  const int wave = tid >> 5;
  const int lane = tid & 31;
  const int sel  = (lane >> 4) & 1;
  const int l16  = lane & 15;
  const int rowBase = blockIdx.x * (WAVES * 16) + wave * 16;

  // --- async-stage this wave's 16 A-rows (hi+lo) into LDS ------------------
  {
    unsigned long long ghi = (unsigned long long)(Ahi + (size_t)rowBase * DDIM);
    unsigned long long glo = (unsigned long long)(Alo + (size_t)rowBase * DDIM);
    unsigned shi = lds_off_of(sAhi[wave]);
    unsigned slo = lds_off_of(sAlo[wave]);
    for (int i = lane; i < 16 * DDIM / 8; i += 32) {
      async_b128(shi + (unsigned)i * 16u, ghi + (unsigned long long)i * 16u);
      async_b128(slo + (unsigned)i * 16u, glo + (unsigned long long)i * 16u);
    }
  }

  // --- async-stage a 32-column B tile into buffer `buf` --------------------
  auto issueB = [&](int buf, int colBase) {
    unsigned long long ghi = (unsigned long long)(Ahi + (size_t)colBase * DDIM);
    unsigned long long glo = (unsigned long long)(Alo + (size_t)colBase * DDIM);
    unsigned shi = lds_off_of(sBhi[buf]);
    unsigned slo = lds_off_of(sBlo[buf]);
    for (int i = tid; i < TILEC * DDIM / 8; i += 32 * WAVES) {
      async_b128(shi + (unsigned)i * 16u, ghi + (unsigned long long)i * 16u);
      async_b128(slo + (unsigned)i * 16u, glo + (unsigned long long)i * 16u);
    }
  };

  issueB(0, 0);
  wait_async0();      // A rows + B tile 0 landed (ASYNCcnt -> 0)
  __syncthreads();

  const float sqi = sq[rowBase + l16];
  float b0 = 3.4e38f, b1 = 3.4e38f, b2 = 3.4e38f;  // top-3 smallest dist^2
  int   i0 = 0, i1 = 0, i2 = 0;

  const _Float16* arowh = sAhi[wave] + l16 * DDIM;
  const _Float16* arowl = sAlo[wave] + l16 * DDIM;

  for (int ct = 0; ct < NT; ++ct) {
    const int colBase = ct * TILEC;
    const int buf = ct & 1;

    // Kick off next tile's copy into the other buffer; overlap with compute.
    if (ct + 1 < NT) {
      issueB(buf ^ 1, colBase + TILEC);
      if (ct + 2 < NT) {    // warm L2 for the tile after that
        const char* nxt = (const char*)(Ahi + (size_t)(colBase + 2 * TILEC) * DDIM);
        __builtin_prefetch(nxt + (size_t)tid * 256, 0, 1);
      }
    }

    // Per-lane column norms for the two sub-tiles (hides latency under WMMA).
    const float sqc0 = sq[colBase + l16];
    const float sqc1 = sq[colBase + 16 + l16];

    // C0/C1 = A_hi B_hi + A_hi B_lo + A_lo B_hi   (fp32 accumulate)
    const _Float16* brh0 = sBhi[buf] + l16 * DDIM;
    const _Float16* brl0 = sBlo[buf] + l16 * DDIM;
    const _Float16* brh1 = brh0 + 16 * DDIM;
    const _Float16* brl1 = brl0 + 16 * DDIM;
    v8f c0 = {0.f, 0.f, 0.f, 0.f, 0.f, 0.f, 0.f, 0.f};
    v8f c1 = {0.f, 0.f, 0.f, 0.f, 0.f, 0.f, 0.f, 0.f};
#pragma unroll
    for (int kk = 0; kk < DDIM; kk += 32) {
      v16h ah  = ld_frag(arowh + kk, sel);
      v16h al  = ld_frag(arowl + kk, sel);
      v16h bh0 = ld_frag(brh0 + kk, sel);
      v16h bl0 = ld_frag(brl0 + kk, sel);
      v16h bh1 = ld_frag(brh1 + kk, sel);
      v16h bl1 = ld_frag(brl1 + kk, sel);
      c0 = __builtin_amdgcn_wmma_f32_16x16x32_f16(false, ah, false, bh0,
                                                  (short)0, c0, false, false);
      c0 = __builtin_amdgcn_wmma_f32_16x16x32_f16(false, ah, false, bl0,
                                                  (short)0, c0, false, false);
      c0 = __builtin_amdgcn_wmma_f32_16x16x32_f16(false, al, false, bh0,
                                                  (short)0, c0, false, false);
      c1 = __builtin_amdgcn_wmma_f32_16x16x32_f16(false, ah, false, bh1,
                                                  (short)0, c1, false, false);
      c1 = __builtin_amdgcn_wmma_f32_16x16x32_f16(false, ah, false, bl1,
                                                  (short)0, c1, false, false);
      c1 = __builtin_amdgcn_wmma_f32_16x16x32_f16(false, al, false, bh1,
                                                  (short)0, c1, false, false);
    }

    // Tile-wide lower bound: min over all rows/cols of (sq[j] - 2*S[i][j]).
    float vm = sqc0 - 2.0f * c0[0];
#pragma unroll
    for (int r = 1; r < 8; ++r) vm = fminf(vm, sqc0 - 2.0f * c0[r]);
#pragma unroll
    for (int r = 0; r < 8; ++r) vm = fminf(vm, sqc1 - 2.0f * c1[r]);
#pragma unroll
    for (int off = 16; off > 0; off >>= 1)
      vm = fminf(vm, __shfl_xor(vm, off, 32));   // tile min in all lanes

    // Scatter C to per-wave LDS scratch: VGPR r / lane -> row (r+sel*8).
#pragma unroll
    for (int r = 0; r < 8; ++r) {
      sS[wave][r + sel * 8][l16]      = c0[r];
      sS[wave][r + sel * 8][16 + l16] = c1[r];
    }
    // Intra-wave only: LDS ops of one wave execute in issue order; a compiler
    // scheduling fence is sufficient (no workgroup barrier needed).
    __builtin_amdgcn_wave_barrier();

    // Lanes 0..15: lane owns one row. Conservative skip: the tile cannot beat
    // this row's current 3rd-best unless sqi + tileMin < b2.
    if (lane < 16 && (sqi + vm) < b2) {
      for (int j = 0; j < TILEC; ++j) {
        const int colj = colBase + j;
        const float d2 = sqi + sq[colj] - 2.0f * sS[wave][lane][j];
        if (d2 < b2) {
          if (d2 < b1) {
            b2 = b1; i2 = i1;
            if (d2 < b0) { b1 = b0; i1 = i0; b0 = d2; i0 = colj; }
            else         { b1 = d2; i1 = colj; }
          } else { b2 = d2; i2 = colj; }
        }
      }
    }

    wait_async0();     // next tile's async writes have landed in LDS
    __syncthreads();   // all waves done with `buf`; next tile may overwrite it
  }

  // i2 == idx3[:,2] (self distance is always rank 0). Share per wave.
  if (lane < 16) sNeg[wave][lane] = i2;
  __syncthreads();

  // Epilogue: d_ap, d_an, per-row loss term (wave-shuffle reduction).
  for (int r = 0; r < 16; ++r) {
    const int row = rowBase + r;
    const int neg = sNeg[wave][r];
    const float* xr = x   + (size_t)row * DDIM;
    const float* pr = pos + (size_t)row * DDIM;
    const float* nr = x   + (size_t)neg * DDIM;
    float accp = 0.f, accn = 0.f;
    for (int k = lane; k < DDIM; k += 32) {
      float xv = xr[k];
      float dp = xv - pr[k] + EPS_F;
      float dn = xv - nr[k] + EPS_F;
      accp += dp * dp;
      accn += dn * dn;
    }
    for (int off = 16; off > 0; off >>= 1) {
      accp += __shfl_down(accp, off, 32);
      accn += __shfl_down(accn, off, 32);
    }
    if (lane == 0)
      lossterm[row] = fmaxf(sqrtf(accp) - sqrtf(accn) + MARGIN_F, 0.f);
  }
}

// ---------------------------------------------------------------------------
// Kernel 3: deterministic mean reduction
// ---------------------------------------------------------------------------
__global__ __launch_bounds__(256) void reduce_kernel(
    const float* __restrict__ t, float* __restrict__ out)
{
  __shared__ float red[256];
  float acc = 0.f;
  for (int i = threadIdx.x; i < NROWS; i += 256) acc += t[i];
  red[threadIdx.x] = acc;
  __syncthreads();
  for (int s = 128; s > 0; s >>= 1) {
    if (threadIdx.x < s) red[threadIdx.x] += red[threadIdx.x + s];
    __syncthreads();
  }
  if (threadIdx.x == 0) out[0] = red[0] * (1.0f / NROWS);
}

// ---------------------------------------------------------------------------
extern "C" void kernel_launch(void* const* d_in, const int* in_sizes, int n_in,
                              void* d_out, int out_size, void* d_ws, size_t ws_size,
                              hipStream_t stream)
{
  const float* x   = (const float*)d_in[0];   // inputs  (N x D) fp32
  const float* pos = (const float*)d_in[1];   // positive(N x D) fp32

  // Workspace layout: f16 hi (16 MB) | f16 lo (16 MB) | sq (64 KB) | loss (64 KB)
  _Float16* Ahi = (_Float16*)d_ws;
  _Float16* Alo = Ahi + (size_t)NROWS * DDIM;
  float* sq       = (float*)(Alo + (size_t)NROWS * DDIM);
  float* lossterm = sq + NROWS;

  prep_kernel<<<NROWS, 128, 0, stream>>>(x, Ahi, Alo, sq);
  triphard_main<<<NROWS / (WAVES * 16), 32 * WAVES, 0, stream>>>(
      x, pos, Ahi, Alo, sq, lossterm);
  reduce_kernel<<<1, 256, 0, stream>>>(lossterm, (float*)d_out);
}
